// GraphAttention_56951266345697
// MI455X (gfx1250) — matrix operand order repaired and checked by reading, not compiled, not run on hardware
//
#include <hip/hip_runtime.h>
#include <hip/hip_bf16.h>

typedef __attribute__((ext_vector_type(16))) __bf16 v16bf;
typedef __attribute__((ext_vector_type(8)))  float  v8f;
typedef __attribute__((ext_vector_type(4)))  unsigned int v4u;
typedef __attribute__((ext_vector_type(8)))  int v8i_t;
typedef __attribute__((ext_vector_type(4)))  int v4i_t;

#define WMMA_BF16(A, B, C) \
  __builtin_amdgcn_wmma_f32_16x16x32_bf16(false, (A), false, (B), (short)0, (C), false, false)

// padded LDS row strides (elements) chosen so TDM pad fields give them
#define KB_STRIDE 136   // 128 + 8  (pad 4 DWORDs per 64-DWORD row)
#define VB_STRIDE 72    // 64 + 8   (pad 4 DWORDs per 32-DWORD row)
#define KB_ELEMS  (64 * KB_STRIDE)   // 8704 elems / buffer
#define VB_ELEMS  (128 * VB_STRIDE)  // 9216 elems / buffer

// ---- vectorized 16 x bf16 fragment loads (force b128 / ds_load_b128) ------
__device__ inline v16bf ld16(const __bf16* p) {           // 16 contiguous
  union { uint4 u[2]; v16bf v; } t;
  t.u[0] = *(const uint4*)p;
  t.u[1] = *(const uint4*)(p + 8);
  return t.v;
}
__device__ inline v16bf ld16_split(const __bf16* p) {     // 8 @p, 8 @p+16
  union { uint4 u[2]; v16bf v; } t;
  t.u[0] = *(const uint4*)p;
  t.u[1] = *(const uint4*)(p + 16);
  return t.v;
}

// ---- Tensor Data Mover: 2-D tile (rows x tile0 elems, 2-byte data) --------
__device__ inline unsigned rfl(unsigned x) {
  return (unsigned)__builtin_amdgcn_readfirstlane((int)x);
}
__device__ inline void tdm_load_2d(unsigned lds_off, unsigned long long ga,
                                   unsigned tile0, unsigned tile1,
                                   unsigned stride0,
                                   unsigned pad_int, unsigned pad_amt)
{
  const unsigned td0 = 1u << 20, td1 = 1u << 20;  // huge dims: no OOB clamping
  v4u g0;
  g0[0] = rfl(1u);                                        // count=1
  g0[1] = rfl(lds_off);                                   // LDS byte offset
  g0[2] = rfl((unsigned)ga);
  g0[3] = rfl((unsigned)((ga >> 32) & 0x01ffffffu) | 0x80000000u); // type=2
  v8i_t g1;
  g1[0] = (int)rfl((1u << 16) | (1u << 20) | (pad_int << 22) | (pad_amt << 25));
  g1[1] = (int)rfl((td0 & 0xffffu) << 16);                // tensor_dim0 lo
  g1[2] = (int)rfl(((td0 >> 16) & 0xffffu) | ((td1 & 0xffffu) << 16));
  g1[3] = (int)rfl(((td1 >> 16) & 0xffffu) | ((tile0 & 0xffffu) << 16));
  g1[4] = (int)rfl(tile1 & 0xffffu);                      // tile_dim1, tile_dim2=0
  g1[5] = (int)rfl(stride0);                              // dim0 stride lo
  g1[6] = 0;
  g1[7] = 0;
  v4i_t z4 = {0, 0, 0, 0};
#if defined(__clang_major__) && __clang_major__ >= 23
  v8i_t z8 = {0, 0, 0, 0, 0, 0, 0, 0};
  __builtin_amdgcn_tensor_load_to_lds(g0, g1, z4, z4, z8, 0);
#else
  __builtin_amdgcn_tensor_load_to_lds(g0, g1, z4, z4, 0);
#endif
}
__device__ inline unsigned lds_off_of(const void* p) {
  return (unsigned)(uintptr_t)p;   // shared aperture: low 32 bits = LDS offset
}

// ---------------------------------------------------------------------------
// f32 -> bf16 conversion (weights; one-shot)
// ---------------------------------------------------------------------------
__global__ __launch_bounds__(256) void ga_cvt_kernel(
    const float* __restrict__ src, __bf16* __restrict__ dst, int n)
{
  int i = blockIdx.x * 256 + threadIdx.x;
  if (i < n) dst[i] = (__bf16)src[i];
}

// ---------------------------------------------------------------------------
// Projection: Y = X @ W^T + bias, optional L2-normalize rows, bf16 out.
// ---------------------------------------------------------------------------
__global__ __launch_bounds__(128, 1) void ga_proj_kernel(
    const float* __restrict__ X, const __bf16* __restrict__ Wb,
    const float* __restrict__ bias, __bf16* __restrict__ Y,
    int K, int N_total, int do_norm, int transpose_out)
{
  const int lane = threadIdx.x & 31;
  const int lid  = lane & 15;
  const int half = lane >> 4;
  const int wave = threadIdx.x >> 5;
  const int r0   = (blockIdx.x * 4 + wave) * 16;
  const int n0   = blockIdx.y * 128;

  v8f c[8];
  #pragma unroll
  for (int nt = 0; nt < 8; ++nt) c[nt] = v8f{};

  const float* xrow = X + (size_t)(r0 + lid) * K;

  const int nkt = K >> 5;
  for (int kt = 0; kt < nkt; ++kt) {
    float af[16];
    {
      const float* p0 = xrow + kt * 32 + half * 8;
      union { uint4 u[2]; float f[8]; } t0, t1;
      t0.u[0] = *(const uint4*)(p0);      t0.u[1] = *(const uint4*)(p0 + 4);
      t1.u[0] = *(const uint4*)(p0 + 16); t1.u[1] = *(const uint4*)(p0 + 20);
      #pragma unroll
      for (int j = 0; j < 8; ++j) { af[j] = t0.f[j]; af[8 + j] = t1.f[j]; }
    }
    v16bf bf[8];
    #pragma unroll
    for (int nt = 0; nt < 8; ++nt)
      bf[nt] = ld16_split(Wb + (size_t)(n0 + nt * 16 + lid) * K + kt * 32 + half * 8);
    v16bf a;
    #pragma unroll
    for (int j = 0; j < 16; ++j) a[j] = (__bf16)af[j];
    #pragma unroll
    for (int nt = 0; nt < 8; ++nt) c[nt] = WMMA_BF16(a, bf[nt], c[nt]);
  }

  #pragma unroll
  for (int nt = 0; nt < 8; ++nt) {
    float bi = bias[n0 + nt * 16 + lid];
    #pragma unroll
    for (int j = 0; j < 8; ++j) c[nt][j] += bi;
  }

  if (do_norm) {
    #pragma unroll
    for (int j = 0; j < 8; ++j) {
      float ss = 0.0f;
      #pragma unroll
      for (int nt = 0; nt < 8; ++nt) ss += c[nt][j] * c[nt][j];
      #pragma unroll
      for (int m = 1; m < 16; m <<= 1) ss += __shfl_xor(ss, m, 32);
      float sc = 1.0f / fmaxf(sqrtf(ss), 1e-12f);
      #pragma unroll
      for (int nt = 0; nt < 8; ++nt) c[nt][j] *= sc;
    }
  }

  #pragma unroll
  for (int j = 0; j < 8; ++j) {
    int row = r0 + j + 8 * half;
    #pragma unroll
    for (int nt = 0; nt < 8; ++nt) {
      int col = n0 + nt * 16 + lid;
      size_t idx;
      if (transpose_out)
        idx = ((size_t)(row & 3) * N_total + col) * 4096 + (size_t)(row >> 2);
      else
        idx = (size_t)row * N_total + col;
      Y[idx] = (__bf16)c[nt][j];
    }
  }
}

// ---------------------------------------------------------------------------
// Flash attention with TDM double-buffered tiles.
// Workgroup = 16 q rows x one batch; 4 waves, each owns a 128-wide D slice.
// s-tile = 64: per iteration 2 tensor_load_to_lds + 32 WMMAs.
// ---------------------------------------------------------------------------
__global__ __launch_bounds__(128, 1) void ga_attn_kernel(
    const __bf16* __restrict__ wq, const __bf16* __restrict__ wk,
    const __bf16* __restrict__ wvt, float* __restrict__ out)
{
  __shared__ __bf16 kbuf[4 * 2 * KB_ELEMS];   // per-wave double-buffered wk tiles
  __shared__ __bf16 vbuf[4 * 2 * VB_ELEMS];   // per-wave double-buffered wv tiles
  __shared__ __bf16 plds[4 * 16 * 64];        // P handoff (C-layout -> A-layout)

  const int lane = threadIdx.x & 31;
  const int lid  = lane & 15;
  const int half = lane >> 4;
  const int wave = threadIdx.x >> 5;
  const int q0   = blockIdx.x * 16;
  const int b    = blockIdx.y;
  const int d0   = wave * 128;
  __bf16* kb0 = kbuf + wave * 2 * KB_ELEMS;
  __bf16* vb0 = vbuf + wave * 2 * VB_ELEMS;
  __bf16* myP = plds + wave * 16 * 64;

  const __bf16* gk = wk + (size_t)b * 128;                 // + s*512
  const __bf16* gv = wvt + ((size_t)b * 512 + d0) * 4096;  // + d*4096 + s

  // wq A fragments (register-resident)
  v16bf aq[4];
  {
    const __bf16* qb = wq + ((size_t)(q0 + lid) * 4 + b) * 128;
    #pragma unroll
    for (int kt = 0; kt < 4; ++kt)
      aq[kt] = ld16_split(qb + kt * 32 + half * 8);
  }

  float mrow[8], lrow[8];
  v8f   o[8];
  #pragma unroll
  for (int j = 0; j < 8; ++j) { mrow[j] = -1e30f; lrow[j] = 0.0f; }
  #pragma unroll
  for (int nt = 0; nt < 8; ++nt) o[nt] = v8f{};

  // prologue: tiles for iterations 0 and 1
  #pragma unroll
  for (int it = 0; it < 2; ++it) {
    tdm_load_2d(lds_off_of(kb0 + it * KB_ELEMS),
                (unsigned long long)(uintptr_t)(gk + (size_t)it * 64 * 512),
                128, 64, 512, 5, 3);
    tdm_load_2d(lds_off_of(vb0 + it * VB_ELEMS),
                (unsigned long long)(uintptr_t)(gv + (size_t)it * 64),
                64, 128, 4096, 4, 3);
  }

  for (int it = 0; it < 64; ++it) {
    const int buf = it & 1;
    // current tiles done; next iteration's 2 loads may remain in flight
    if (it < 63) __builtin_amdgcn_s_wait_tensorcnt((short)2);
    else         __builtin_amdgcn_s_wait_tensorcnt((short)0);

    const __bf16* kb = kb0 + buf * KB_ELEMS;
    const __bf16* vb = vb0 + buf * VB_ELEMS;

    // ============ phase A: scores (4 C tiles over s-tile of 64) ============
    v8f S[4];
    #pragma unroll
    for (int t = 0; t < 4; ++t) {
      const __bf16* kr = kb + (t * 16 + lid) * KB_STRIDE;
      v16bf bkf[4];
      #pragma unroll
      for (int kt = 0; kt < 4; ++kt)
        bkf[kt] = ld16_split(kr + kt * 32 + half * 8);
      S[t] = v8f{};
      #pragma unroll
      for (int kt = 0; kt < 4; ++kt) S[t] = WMMA_BF16(aq[kt], bkf[kt], S[t]);
    }

    // ============ phase B: online softmax (scale 1/TAU = 30) ===============
    #pragma unroll
    for (int j = 0; j < 8; ++j) {
      float x0 = S[0][j] * 30.0f, x1 = S[1][j] * 30.0f;
      float x2 = S[2][j] * 30.0f, x3 = S[3][j] * 30.0f;
      float tm = fmaxf(fmaxf(x0, x1), fmaxf(x2, x3));
      #pragma unroll
      for (int m = 1; m < 16; m <<= 1) tm = fmaxf(tm, __shfl_xor(tm, m, 32));
      float mn    = fmaxf(mrow[j], tm);
      float alpha = __expf(mrow[j] - mn);
      float p0 = __expf(x0 - mn), p1 = __expf(x1 - mn);
      float p2 = __expf(x2 - mn), p3 = __expf(x3 - mn);
      float rs = (p0 + p1) + (p2 + p3);
      #pragma unroll
      for (int m = 1; m < 16; m <<= 1) rs += __shfl_xor(rs, m, 32);
      lrow[j] = lrow[j] * alpha + rs;
      mrow[j] = mn;
      #pragma unroll
      for (int nt = 0; nt < 8; ++nt) o[nt][j] *= alpha;
      __bf16* pr = myP + (j + 8 * half) * 64 + lid;
      pr[0]  = (__bf16)p0;
      pr[16] = (__bf16)p1;
      pr[32] = (__bf16)p2;
      pr[48] = (__bf16)p3;
    }
    asm volatile("s_wait_dscnt 0x0" ::: "memory");

    // ============ phase C: O += P @ wv (two K=32 halves) ===================
    #pragma unroll
    for (int sh = 0; sh < 2; ++sh) {
      v16bf ap = ld16_split(myP + lid * 64 + sh * 32 + half * 8);
      v16bf bv[8];
      #pragma unroll
      for (int nt = 0; nt < 8; ++nt)
        bv[nt] = ld16_split(vb + (nt * 16 + lid) * VB_STRIDE + sh * 32 + half * 8);
      #pragma unroll
      for (int nt = 0; nt < 8; ++nt) o[nt] = WMMA_BF16(ap, bv[nt], o[nt]);
    }

    // ============ refill the buffer just consumed with tiles for it+2 ======
    if (it + 2 < 64) {
      asm volatile("s_wait_dscnt 0x0" ::: "memory");  // LDS reads drained
      tdm_load_2d(lds_off_of(kb0 + buf * KB_ELEMS),
                  (unsigned long long)(uintptr_t)(gk + (size_t)(it + 2) * 64 * 512),
                  128, 64, 512, 5, 3);
      tdm_load_2d(lds_off_of(vb0 + buf * VB_ELEMS),
                  (unsigned long long)(uintptr_t)(gv + (size_t)(it + 2) * 64),
                  64, 128, 4096, 4, 3);
    }
  }

  // ============ finalize: divide by row sums, store f32 ====================
  #pragma unroll
  for (int j = 0; j < 8; ++j) {
    float inv = 1.0f / lrow[j];
    int q = q0 + j + 8 * half;
    #pragma unroll
    for (int nt = 0; nt < 8; ++nt)
      out[((size_t)q * 4 + b) * 512 + d0 + nt * 16 + lid] = o[nt][j] * inv;
  }
}

// ---------------------------------------------------------------------------
extern "C" void kernel_launch(void* const* d_in, const int* in_sizes, int n_in,
                              void* d_out, int out_size, void* d_ws, size_t ws_size,
                              hipStream_t stream) {
  const float* query = (const float*)d_in[0];
  const float* k_in  = (const float*)d_in[1];
  const float* value = (const float*)d_in[2];
  const float* WK_w  = (const float*)d_in[3];
  const float* WK_b  = (const float*)d_in[4];
  const float* WV_w  = (const float*)d_in[5];
  const float* WV_b  = (const float*)d_in[6];
  float* out = (float*)d_out;

  __bf16* wq   = (__bf16*)d_ws;                      // 16384*128
  __bf16* wk   = wq  + (size_t)16384 * 128;          // 16384*128
  __bf16* wvt  = wk  + (size_t)16384 * 128;          // 4*512*4096
  __bf16* wkwb = wvt + (size_t)4 * 512 * 4096;       // 128*512
  __bf16* wvwb = wkwb + (size_t)128 * 512;           // 512*512

  ga_cvt_kernel<<<(128 * 512 + 255) / 256, 256, 0, stream>>>(WK_w, wkwb, 128 * 512);
  ga_cvt_kernel<<<(512 * 512 + 255) / 256, 256, 0, stream>>>(WV_w, wvwb, 512 * 512);

  ga_proj_kernel<<<dim3(256, 1), 128, 0, stream>>>(query, wkwb, WK_b, wq,
                                                   512, 128, 1, 0);
  ga_proj_kernel<<<dim3(256, 1), 128, 0, stream>>>(k_in, wkwb, WK_b, wk,
                                                   512, 128, 1, 0);
  ga_proj_kernel<<<dim3(256, 4), 128, 0, stream>>>(value, wvwb, WV_b, wvt,
                                                   512, 512, 0, 1);
  ga_attn_kernel<<<dim3(256, 4), 128, 0, stream>>>(wq, wk, wvt, out);
}
